// SingelGnn_80719615361505
// MI455X (gfx1250) — matrix operand order repaired and checked by reading, compiled, not run on hardware
//
#include <hip/hip_runtime.h>
#include <hip/hip_bf16.h>
#include <stdint.h>

#define N_NODES 50000
#define N_EDGES 800000
#define HIDDEN  128

typedef __attribute__((ext_vector_type(16))) _Float16 v16h;
typedef __attribute__((ext_vector_type(8)))  float    v8f;

// ---------------------------------------------------------------------------
// Zero-fill (capture-safe replacement for memset; aggr must be 0 every call)
// ---------------------------------------------------------------------------
__global__ void zero_f32(float* __restrict__ p, int n) {
  int i = blockIdx.x * blockDim.x + threadIdx.x;
  int stride = gridDim.x * blockDim.x;
  for (; i < n; i += stride) p[i] = 0.0f;
}

// ---------------------------------------------------------------------------
// Pack W (128x128 f32, row-major [k][n]) into WMMA-B-fragment-native f16:
// layout [ctile(8)][ktile(4)][lane(32)][16 halves], so each B fragment is one
// contiguous 32-byte load per lane.
//   lane <  16: col = ct*16 + lane,      K = kt*32 + j        (j = 0..15)
//   lane >= 16: col = ct*16 + lane - 16, K = kt*32 + 16 + j
// ---------------------------------------------------------------------------
__global__ void pack_w_f16(const float* __restrict__ W, _Float16* __restrict__ Wp) {
  int idx  = blockIdx.x * blockDim.x + threadIdx.x;   // 0 .. 16383
  int j    = idx & 15;
  int lane = (idx >> 4) & 31;
  int kt   = (idx >> 9) & 3;
  int ct   = idx >> 11;
  int col  = ct * 16 + (lane & 15);
  int k    = kt * 32 + (lane >> 4) * 16 + j;
  Wp[idx] = (_Float16)W[k * HIDDEN + col];
}

// ---------------------------------------------------------------------------
// Fused GINE message + scatter: one wave per edge, lane owns 4 columns.
//   msg = relu(x[src] + edge_attr @ We + be);  aggr[dst] += msg (f32 atomics)
// Edge MLP (2->128) recomputed inline: 2 FMAs/element, never materialized.
// ---------------------------------------------------------------------------
__global__ __launch_bounds__(256)
void gine_scatter(const float* __restrict__ x,
                  const long long* __restrict__ edge_index,  // [2][E] int64
                  const float* __restrict__ ea,              // [E][2]
                  const float* __restrict__ We,              // [2][128]
                  const float* __restrict__ be,              // [128]
                  float* __restrict__ aggr) {
  int e    = (blockIdx.x * blockDim.x + threadIdx.x) >> 5;
  int lane = threadIdx.x & 31;
  if (e >= N_EDGES) return;

  long long src = edge_index[e];
  long long dst = edge_index[N_EDGES + e];
  float a0 = ea[2 * e + 0];
  float a1 = ea[2 * e + 1];

  int c = lane * 4;
  float4 xv = *(const float4*)(x  + (size_t)src * HIDDEN + c);
  float4 w0 = *(const float4*)(We + 0 * HIDDEN + c);
  float4 w1 = *(const float4*)(We + 1 * HIDDEN + c);
  float4 bv = *(const float4*)(be + c);

  float m0 = fmaxf(xv.x + fmaf(a0, w0.x, fmaf(a1, w1.x, bv.x)), 0.0f);
  float m1 = fmaxf(xv.y + fmaf(a0, w0.y, fmaf(a1, w1.y, bv.y)), 0.0f);
  float m2 = fmaxf(xv.z + fmaf(a0, w0.z, fmaf(a1, w1.z, bv.z)), 0.0f);
  float m3 = fmaxf(xv.w + fmaf(a0, w0.w, fmaf(a1, w1.w, bv.w)), 0.0f);

  float* ap = aggr + (size_t)dst * HIDDEN + c;
  atomicAdd(ap + 0, m0);
  atomicAdd(ap + 1, m1);
  atomicAdd(ap + 2, m2);
  atomicAdd(ap + 3, m3);
}

// ---------------------------------------------------------------------------
// out = relu(((1+eps)*x + aggr) @ W + b), W pre-packed f16, f32 accumulate.
// Block = 256 threads = 8 waves; wave w owns the 16x16 tile at cols w*16.
// K = 128 covered by 4x v_wmma_f32_16x16x32_f16 per wave.
// ---------------------------------------------------------------------------
__global__ __launch_bounds__(256)
void gine_gemm_relu(const float* __restrict__ x,
                    const float* __restrict__ aggr,
                    const _Float16* __restrict__ Wp,   // packed [8][4][32][16]
                    const float* __restrict__ bias,    // [128]
                    const float* __restrict__ eps_p,   // [1]
                    float* __restrict__ out) {
  // A tile staged as f16, row stride 136 halves (=272B) to spread LDS banks.
  __shared__ _Float16 As[16 * 136];

  const int tid     = threadIdx.x;
  const int rowbase = blockIdx.x * 16;           // 50000 = 3125*16, no tail
  const float s     = 1.0f + eps_p[0];

  // Stage z = s*x + aggr for 16 rows x 128 cols; thread -> (row, 8 cols).
  {
    int r  = tid >> 4;           // 0..15
    int k0 = (tid & 15) * 8;     // 0,8,...,120
    const float4* xp = (const float4*)(x    + (size_t)(rowbase + r) * HIDDEN + k0);
    const float4* gp = (const float4*)(aggr + (size_t)(rowbase + r) * HIDDEN + k0);
    float4 xa = xp[0], xb = xp[1], ga = gp[0], gb = gp[1];
    _Float16* dst = As + r * 136 + k0;
    dst[0] = (_Float16)fmaf(s, xa.x, ga.x);
    dst[1] = (_Float16)fmaf(s, xa.y, ga.y);
    dst[2] = (_Float16)fmaf(s, xa.z, ga.z);
    dst[3] = (_Float16)fmaf(s, xa.w, ga.w);
    dst[4] = (_Float16)fmaf(s, xb.x, gb.x);
    dst[5] = (_Float16)fmaf(s, xb.y, gb.y);
    dst[6] = (_Float16)fmaf(s, xb.z, gb.z);
    dst[7] = (_Float16)fmaf(s, xb.w, gb.w);
  }
  __syncthreads();

  const int wave = tid >> 5;        // 0..7 -> column tile
  const int lane = tid & 31;
  const int half = lane >> 4;       // 0/1
  const int mrow = lane & 15;

  // C/D layout: lane holds col N = wave*16 + mrow for rows (i + 8*half).
  float bcol = bias[wave * 16 + mrow];
  v8f c;
#pragma unroll
  for (int i = 0; i < 8; ++i) c[i] = bcol;

  const uint32_t* lrow = (const uint32_t*)(As + mrow * 136);  // K-pairs as b32

#pragma unroll
  for (int kt = 0; kt < 4; ++kt) {
    // A fragment, contiguous K order (matches packed B):
    //   half==0: K in {kb..kb+15} with kb = kt*32
    //   half==1: same with kb = kt*32 + 16
    union { v16h v; uint32_t u[8]; } a;
    int kb = kt * 32 + half * 16;
#pragma unroll
    for (int j = 0; j < 4; ++j) a.u[j]     = lrow[(kb >> 1) + j];
#pragma unroll
    for (int j = 0; j < 4; ++j) a.u[4 + j] = lrow[(kb >> 1) + 4 + j];

    // B fragment: one contiguous 32B load from the packed layout.
    v16h b = *(const v16h*)(Wp + (size_t)(((wave * 4 + kt) * 32 + lane) * 16));

    c = __builtin_amdgcn_wmma_f32_16x16x32_f16(
        /*neg_a=*/false, a.v, /*neg_b=*/false, b,
        /*c_mod=*/(short)0, c, /*reuse_a=*/false, /*reuse_b=*/false);
  }

  // ReLU + store: row = rowbase + i + 8*half, col = wave*16 + mrow.
  float* op = out + (size_t)(rowbase + half * 8) * HIDDEN + wave * 16 + mrow;
#pragma unroll
  for (int i = 0; i < 8; ++i)
    op[(size_t)i * HIDDEN] = fmaxf(c[i], 0.0f);
}

// ---------------------------------------------------------------------------
extern "C" void kernel_launch(void* const* d_in, const int* in_sizes, int n_in,
                              void* d_out, int out_size, void* d_ws, size_t ws_size,
                              hipStream_t stream) {
  const float*     x    = (const float*)d_in[0];
  const long long* ei   = (const long long*)d_in[1];   // int64 [2][E]
  const float*     ea   = (const float*)d_in[2];
  const float*     We1  = (const float*)d_in[3];
  const float*     be1  = (const float*)d_in[4];
  const float*     W1   = (const float*)d_in[5];
  const float*     b1   = (const float*)d_in[6];
  const float*     eps1 = (const float*)d_in[7];
  const float*     We2  = (const float*)d_in[8];
  const float*     be2  = (const float*)d_in[9];
  const float*     W2   = (const float*)d_in[10];
  const float*     b2   = (const float*)d_in[11];
  const float*     eps2 = (const float*)d_in[12];
  float*           out  = (float*)d_out;

  char* ws = (char*)d_ws;
  float*     aggr = (float*)(ws);                       // 25,600,000 B
  float*     h1   = (float*)(ws + 25600000);            // 25,600,000 B
  _Float16*  Wp1  = (_Float16*)(ws + 51200000);         // 32,768 B
  _Float16*  Wp2  = (_Float16*)(ws + 51232768);         // 32,768 B

  const int NH = N_NODES * HIDDEN;

  // Pre-pack weights into WMMA-B-native f16 layout.
  pack_w_f16<<<64, 256, 0, stream>>>(W1, Wp1);
  pack_w_f16<<<64, 256, 0, stream>>>(W2, Wp2);

  // ---- Layer 1 ----
  zero_f32<<<2048, 256, 0, stream>>>(aggr, NH);
  gine_scatter<<<N_EDGES / 8, 256, 0, stream>>>(x, ei, ea, We1, be1, aggr);
  gine_gemm_relu<<<N_NODES / 16, 256, 0, stream>>>(x, aggr, Wp1, b1, eps1, h1);

  // ---- Layer 2 ----
  zero_f32<<<2048, 256, 0, stream>>>(aggr, NH);
  gine_scatter<<<N_EDGES / 8, 256, 0, stream>>>(h1, ei, ea, We2, be2, aggr);
  gine_gemm_relu<<<N_NODES / 16, 256, 0, stream>>>(h1, aggr, Wp2, b2, eps2, out);
}
